// SampleAndAggregate_28767690949360
// MI455X (gfx1250) — compile-verified, matrix operand order checked by reading, not run on hardware
//
#include <hip/hip_runtime.h>
#include <stdint.h>

// ---------------------------------------------------------------------------
// CDNA5 (gfx1250) wave32 WMMA types
// ---------------------------------------------------------------------------
typedef __attribute__((ext_vector_type(16))) __bf16    v16bf;
typedef __attribute__((ext_vector_type(8)))  float     v8f;
typedef __attribute__((ext_vector_type(8)))  uint32_t  v8u;

__device__ __forceinline__ uint32_t bf16_rne(float f) {
    uint32_t u = __builtin_bit_cast(uint32_t, f);
    return (u + 0x7FFFu + ((u >> 16) & 1u)) >> 16;   // round-to-nearest-even
}
__device__ __forceinline__ uint32_t pack_bf16(float lo, float hi) {
    return bf16_rne(lo) | (bf16_rne(hi) << 16);
}

__device__ __forceinline__ v8f wmma_bf16(v8u a, v8u b, v8f c) {
    return __builtin_amdgcn_wmma_f32_16x16x32_bf16(
        /*neg_a=*/false, __builtin_bit_cast(v16bf, a),
        /*neg_b=*/false, __builtin_bit_cast(v16bf, b),
        /*c_mod=*/(short)0, c, /*reuse_a=*/false, /*reuse_b=*/false);
}

// ---------------------------------------------------------------------------
// Linear f32 -> packed-bf16 conversion: out[i] = pack(in[2i], in[2i+1]).
// Used once for the feature table (rows are contiguous, packing is along K).
// ---------------------------------------------------------------------------
__global__ void convert_bf16(const float* __restrict__ in,
                             uint32_t* __restrict__ out, long n) {
    long i = (long)blockIdx.x * blockDim.x + threadIdx.x;
    if (i >= n) return;
    out[i] = pack_bf16(in[2 * i], in[2 * i + 1]);
}

// ---------------------------------------------------------------------------
// Weight transpose + f32 -> bf16 pack.  W: [K,N] f32  ->  Wt: [N, K/2] u32
// (two bf16 along K per dword, K-major so B fragments load like A fragments)
// ---------------------------------------------------------------------------
__global__ void transpose_to_bf16(const float* __restrict__ W,
                                  uint32_t* __restrict__ Wt, int K, int N) {
    int i = blockIdx.x * blockDim.x + threadIdx.x;
    int kd2 = K >> 1;
    if (i >= N * kd2) return;
    int n  = i / kd2;
    int kk = (i - n * kd2) * 2;
    Wt[i] = pack_bf16(W[(long)kk * N + n], W[(long)(kk + 1) * N + n]);
}

// ---------------------------------------------------------------------------
// Deterministic neighbor sampling: dst[s*num_samp + j] = adj[src[s]*max_deg + j]
// ---------------------------------------------------------------------------
__global__ void gather_neighbors(const int* __restrict__ src,
                                 const int* __restrict__ adj,
                                 int* __restrict__ dst,
                                 int n_src, int num_samp, int max_deg) {
    int i = blockIdx.x * blockDim.x + threadIdx.x;
    if (i >= n_src * num_samp) return;
    int s = i / num_samp, j = i - s * num_samp;
    dst[i] = adj[(long)src[s] * max_deg + j];
}

// ---------------------------------------------------------------------------
// pool_mlp_max<MT>: pooled[g,:] = max_j relu( feat[idx[g*kn+j]] @ Wpool + b )
//   feat rows are packed bf16 (KD2 dwords).  K=256, NOUT=512.  One workgroup
//   (8 waves) per group; MT = # of 16-row M-tiles (compile time).
//   Output written as packed bf16 [groups, NOUT/2 dwords].
// ---------------------------------------------------------------------------
#define KDIM 256
#define KD2  128          // KDIM/2 dwords per row
#define NOUT 512

template <int MT>
__global__ __launch_bounds__(256) void pool_mlp_max(
    const uint32_t* __restrict__ feat, const int* __restrict__ idx,
    const uint32_t* __restrict__ Wt, const float* __restrict__ bias,
    uint32_t* __restrict__ out, int kneigh) {

    __shared__ uint32_t sA[MT * 16 * KD2];   // MT*16 rows x 256 bf16
    const int g   = blockIdx.x;
    const int tid = threadIdx.x;

    // ---- stage gathered bf16 rows into LDS (uint4 = 8 bf16 per copy) ----
    for (int i = tid; i < MT * 16 * (KD2 / 4); i += 256) {
        int r   = i >> 5;                    // / (KD2/4)
        int kd4 = (i & 31) * 4;
        uint4 v = make_uint4(0u, 0u, 0u, 0u);
        if (r < kneigh) {
            long row = idx ? (long)idx[(long)g * kneigh + r] : ((long)g * kneigh + r);
            v = *(const uint4*)(feat + row * KD2 + kd4);
        }
        *(uint4*)(sA + r * KD2 + kd4) = v;
    }
    __syncthreads();

    const int lane = tid & 31;
    const int wave = tid >> 5;
    const int m16  = lane & 15;
    const int kb   = (lane >> 4) * 8;        // half-index offset of this lane half

    for (int nt = wave; nt < NOUT / 16; nt += 8) {
        v8f acc[MT];
        #pragma unroll
        for (int mt = 0; mt < MT; ++mt) acc[mt] = (v8f){};

        const uint32_t* bp = Wt + (long)(nt * 16 + m16) * KD2;
        #pragma unroll
        for (int kt = 0; kt < KDIM / 32; ++kt) {
            const int kh0 = (kt * 32 + kb) >> 1;        // dwords, run 0
            const int kh1 = (kt * 32 + 16 + kb) >> 1;   // dwords, run 1
            v8u b;
            #pragma unroll
            for (int d = 0; d < 4; ++d) { b[d] = bp[kh0 + d]; b[4 + d] = bp[kh1 + d]; }
            #pragma unroll
            for (int mt = 0; mt < MT; ++mt) {
                const uint32_t* ap = sA + (mt * 16 + m16) * KD2;
                v8u a;
                #pragma unroll
                for (int d = 0; d < 4; ++d) { a[d] = ap[kh0 + d]; a[4 + d] = ap[kh1 + d]; }
                acc[mt] = wmma_bf16(a, b, acc[mt]);
            }
        }
        // ---- bias + relu + mask padded rows + max over neighbors ----
        const float bcol  = bias ? bias[nt * 16 + m16] : 0.0f;
        const int   mbase = (lane >> 4) * 8;   // C layout: M = r + (lane>=16 ? 8:0)
        float mx = 0.0f;                       // relu >= 0 -> 0 is a safe identity
        #pragma unroll
        for (int mt = 0; mt < MT; ++mt) {
            #pragma unroll
            for (int r = 0; r < 8; ++r) {
                int   Mrow = mt * 16 + mbase + r;
                float v    = fmaxf(acc[mt][r] + bcol, 0.0f);
                if (Mrow < kneigh) mx = fmaxf(mx, v);
            }
        }
        mx = fmaxf(mx, __shfl_xor(mx, 16, 32));   // combine lane halves (M 0-7 vs 8-15)
        // adjacent columns live in adjacent lanes -> pack pairs, even lanes store
        float mxp = __shfl_xor(mx, 1, 32);
        if (lane < 16 && (lane & 1) == 0)
            out[(long)g * (NOUT / 2) + ((nt * 16 + m16) >> 1)] = pack_bf16(mx, mxp);
    }
}

// ---------------------------------------------------------------------------
// gemm_gather: Out[m, coloff+n] = act( A[idx[m]] (or A[m]) @ Wt^T )
//   A: packed bf16 [*, K/2] u32;  Wt: [NN, K/2] u32.  One wave per 16x16 tile.
//   obf16 = 1 -> output packed bf16 (ldo/2 dwords per row), else f32.
// ---------------------------------------------------------------------------
__global__ __launch_bounds__(256) void gemm_gather(
    const uint32_t* __restrict__ A, const int* __restrict__ idx,
    const uint32_t* __restrict__ Wt, float* __restrict__ out,
    int M, int K, int NN, int ldo, int coloff, int relu, int obf16) {

    const int waveGlobal = (blockIdx.x * 256 + threadIdx.x) >> 5;
    const int lane   = threadIdx.x & 31;
    const int ntiles = NN >> 4;
    const int mtiles = M >> 4;
    if (waveGlobal >= mtiles * ntiles) return;   // wave-uniform exit

    const int mtile = waveGlobal / ntiles;
    const int ntile = waveGlobal - mtile * ntiles;
    const int m16 = lane & 15;
    const int kb  = (lane >> 4) * 8;
    const int kd2 = K >> 1;

    const int  arow  = mtile * 16 + m16;
    const long arowg = idx ? (long)idx[arow] : (long)arow;
    const uint32_t* ap = A  + arowg * kd2;
    const uint32_t* bp = Wt + (long)(ntile * 16 + m16) * kd2;

    v8f acc = {};
    for (int kt = 0; kt < (K >> 5); ++kt) {
        const int kh0 = (kt * 32 + kb) >> 1;
        const int kh1 = (kt * 32 + 16 + kb) >> 1;
        v8u a, b;
        #pragma unroll
        for (int d = 0; d < 4; ++d) {
            a[d] = ap[kh0 + d];  a[4 + d] = ap[kh1 + d];
            b[d] = bp[kh0 + d];  b[4 + d] = bp[kh1 + d];
        }
        acc = wmma_bf16(a, b, acc);
    }
    const int mb = (lane >> 4) * 8;
    if (obf16) {
        uint32_t* o32 = (uint32_t*)out;
        #pragma unroll
        for (int r = 0; r < 8; ++r) {
            float v = acc[r];
            if (relu) v = fmaxf(v, 0.0f);
            float vp = __shfl_xor(v, 1, 32);     // partner column (lane^1)
            if ((m16 & 1) == 0)
                o32[(long)(mtile * 16 + mb + r) * (ldo >> 1) +
                    ((coloff + ntile * 16 + m16) >> 1)] = pack_bf16(v, vp);
        }
    } else {
        #pragma unroll
        for (int r = 0; r < 8; ++r) {
            float v = acc[r];
            if (relu) v = fmaxf(v, 0.0f);
            out[(long)(mtile * 16 + mb + r) * ldo + coloff + ntile * 16 + m16] = v;
        }
    }
}

// ---------------------------------------------------------------------------
// Row L2-normalize: out[row] = in[row] / max(||in[row]||, 1e-12).  cols == 256.
// ---------------------------------------------------------------------------
__global__ __launch_bounds__(256) void normalize_rows(
    const float* __restrict__ in, float* __restrict__ out, int cols) {
    __shared__ float red[256];
    const int row = blockIdx.x, tid = threadIdx.x;
    float v = in[(long)row * cols + tid];
    red[tid] = v * v;
    __syncthreads();
    for (int off = 128; off > 0; off >>= 1) {
        if (tid < off) red[tid] += red[tid + off];
        __syncthreads();
    }
    float norm = fmaxf(sqrtf(red[0]), 1e-12f);
    out[(long)row * cols + tid] = v / norm;
}

// ---------------------------------------------------------------------------
// Host orchestration
// ---------------------------------------------------------------------------
static inline void launch_gemm(const uint32_t* A, const int* idx, const uint32_t* Wt,
                               void* out, int M, int K, int NN, int ldo, int coloff,
                               int relu, int obf16, hipStream_t stream) {
    int tiles  = (M / 16) * (NN / 16);
    int blocks = (tiles + 7) / 8;            // 8 waves per 256-thread block
    gemm_gather<<<blocks, 256, 0, stream>>>(A, idx, Wt, (float*)out,
                                            M, K, NN, ldo, coloff, relu, obf16);
}

extern "C" void kernel_launch(void* const* d_in, const int* in_sizes, int n_in,
                              void* d_out, int out_size, void* d_ws, size_t ws_size,
                              hipStream_t stream) {
    const int*   batch1   = (const int*)d_in[0];
    const int*   batch2   = (const int*)d_in[1];
    const float* features = (const float*)d_in[2];
    const int*   adj      = (const int*)d_in[3];
    const float* W_pool0  = (const float*)d_in[4];
    const float* b_pool0  = (const float*)d_in[5];
    const float* W_neigh0 = (const float*)d_in[6];
    const float* W_self0  = (const float*)d_in[7];
    const float* W_pool1  = (const float*)d_in[8];
    const float* b_pool1  = (const float*)d_in[9];
    const float* W_neigh1 = (const float*)d_in[10];
    const float* W_self1  = (const float*)d_in[11];

    const int B         = in_sizes[0];                 // 512
    const int feat_dim  = 256;
    const int num_nodes = in_sizes[2] / feat_dim;
    const int max_deg   = in_sizes[3] / num_nodes;     // 128
    const int NS0 = 25, NS1 = 10;                      // fanouts
    const int dcat = 256;                              // 2*d_out

    // ---- workspace carve (aligned 256B) ----
    char* p = (char*)d_ws;
    auto carve = [&](size_t bytes) { char* r = p; p += (bytes + 255) & ~(size_t)255; return r; };
    uint32_t* featbf    = (uint32_t*)carve((size_t)num_nodes * KD2 * 4);  // bf16 features
    uint32_t* Wt_pool0  = (uint32_t*)carve((size_t)512 * 128 * 4);  // [512,256] bf16
    uint32_t* Wt_neigh0 = (uint32_t*)carve((size_t)128 * 256 * 4);  // [128,512] bf16
    uint32_t* Wt_self0  = (uint32_t*)carve((size_t)128 * 128 * 4);  // [128,256] bf16
    uint32_t* Wt_pool1  = (uint32_t*)carve((size_t)512 * 128 * 4);
    uint32_t* Wt_neigh1 = (uint32_t*)carve((size_t)128 * 256 * 4);
    uint32_t* Wt_self1  = (uint32_t*)carve((size_t)128 * 128 * 4);
    int*      n1      = (int*)carve((size_t)B * NS1 * 4);               // 5120
    int*      n2      = (int*)carve((size_t)B * NS1 * NS0 * 4);         // 128000
    uint32_t* pooled1 = (uint32_t*)carve((size_t)B * NS1 * (NOUT/2) * 4); // [5120,512] bf16
    uint32_t* pooled0 = (uint32_t*)carve((size_t)B * (NOUT/2) * 4);       // [512,512] bf16
    uint32_t* h1      = (uint32_t*)carve((size_t)B * NS1 * (dcat/2) * 4); // [5120,256] bf16
    uint32_t* h0      = (uint32_t*)carve((size_t)B * (dcat/2) * 4);       // [512,256] bf16
    float*    fin     = (float*)carve((size_t)B * dcat * 4);              // [512,256] f32

    // ---- one-time conversions ----
    {
        long n = (long)num_nodes * KD2;          // packed dwords in feature table
        convert_bf16<<<(int)((n + 255) / 256), 256, 0, stream>>>(features, featbf, n);
    }
    auto tpose = [&](const float* W, uint32_t* Wt, int K, int N) {
        int total = N * (K / 2);
        transpose_to_bf16<<<(total + 255) / 256, 256, 0, stream>>>(W, Wt, K, N);
    };
    tpose(W_pool0,  Wt_pool0,  256, 512);
    tpose(W_neigh0, Wt_neigh0, 512, 128);
    tpose(W_self0,  Wt_self0,  256, 128);
    tpose(W_pool1,  Wt_pool1,  256, 512);
    tpose(W_neigh1, Wt_neigh1, 512, 128);
    tpose(W_self1,  Wt_self1,  256, 128);

    for (int bi = 0; bi < 2; ++bi) {
        const int* batch = bi ? batch2 : batch1;
        float* outp = (float*)d_out + (size_t)bi * B * dcat;

        // sampling
        gather_neighbors<<<(B * NS1 + 255) / 256, 256, 0, stream>>>(batch, adj, n1, B, NS1, max_deg);
        gather_neighbors<<<(B * NS1 * NS0 + 255) / 256, 256, 0, stream>>>(n1, adj, n2, B * NS1, NS0, max_deg);

        // layer 0 pooled-MLP + maxpool (hot path: 128000 x [256->512] bf16 WMMA)
        pool_mlp_max<2><<<B * NS1, 256, 0, stream>>>(featbf, n2, Wt_pool0, b_pool0, pooled1, NS0);
        pool_mlp_max<1><<<B,       256, 0, stream>>>(featbf, n1, Wt_pool0, b_pool0, pooled0, NS1);

        // layer 0 outputs: h = relu([self@Wself | pooled@Wneigh])  (bf16 out)
        launch_gemm(featbf,  n1,      Wt_self0,  h1, B * NS1, 256, 128, dcat, 0,   1, 1, stream);
        launch_gemm(pooled1, nullptr, Wt_neigh0, h1, B * NS1, 512, 128, dcat, 128, 1, 1, stream);
        launch_gemm(featbf,  batch,   Wt_self0,  h0, B,       256, 128, dcat, 0,   1, 1, stream);
        launch_gemm(pooled0, nullptr, Wt_neigh0, h0, B,       512, 128, dcat, 128, 1, 1, stream);

        // layer 1 (identity act); neighbors of group b are h1 rows b*10..b*10+9
        pool_mlp_max<1><<<B, 256, 0, stream>>>(h1, nullptr, Wt_pool1, b_pool1, pooled0, NS1);
        launch_gemm(h0,      nullptr, Wt_self1,  fin, B, 256, 128, dcat, 0,   0, 0, stream);
        launch_gemm(pooled0, nullptr, Wt_neigh1, fin, B, 512, 128, dcat, 128, 0, 0, stream);

        normalize_rows<<<B, 256, 0, stream>>>(fin, outp, dcat);
    }
}
